// Community_29575144800945
// MI455X (gfx1250) — matrix operand order repaired and checked by reading, not compiled, not run on hardware
//
#include <hip/hip_runtime.h>
#include <hip/hip_bf16.h>

// ---------------------------------------------------------------------------
// Modular RNN (community RNN) for MI455X / gfx1250, wave32 + WMMA bf16.
//   HX[t] = [x_t | h_{t-1}]  (B x 2560, bf16)
//   h_t = tanh(HX[t] @ Wcat^T + (b_ih+b_hh)) -> written into HX[t+1] h-columns
//   out[t,m,b,o] = h_t[b,:] @ Wout_m^T + b_out   (block-diagonal: K=512/module)
// GEMMs: v_wmma_f32_16x16x32_bf16 (f32 accumulate).
// Step kernel: 32 WGs x 8 waves; wave = 16 batch rows x 64 hidden cols (4 tiles).
// Wcat chunks staged to double-buffered LDS via GLOBAL_LOAD_ASYNC_TO_LDS_B128
// (ASYNCcnt path; AS3 dest pointers pre-cast outside the loop), else
// load+ds_store fallback. 1 barrier / K64 chunk; branch-free inner loop.
// Workspace requirement: ~77 MB.
// ---------------------------------------------------------------------------

typedef __bf16 bf16;
typedef __attribute__((ext_vector_type(16))) bf16         v16bf;
typedef __attribute__((ext_vector_type(8)))  float        v8f;
typedef __attribute__((ext_vector_type(4)))  unsigned int u32x4;
typedef __attribute__((ext_vector_type(4)))  int          i32x4;

#define MMOD   4
#define HID    512
#define HTOT   2048      // MMOD*HID
#define INTOT  512       // MMOD*128
#define OUTTOT 64        // MMOD*16
#define TSTEPS 100
#define BATCH  128
#define KCAT   2560      // INTOT + HTOT

#define NBLK   64        // hidden cols per workgroup
#define NTILES 4         // N tiles per wave
#define KCH    64        // staged K chunk width (2 wmma sub-steps)
#define NCHUNK (KCAT / KCH)       // 40
#define LROW   72        // bf16 elems per LDS row (144 B; 36 banks -> conflict-free b128)

#ifndef __has_builtin
#define __has_builtin(x) 0
#endif

#if __has_builtin(__builtin_amdgcn_global_load_async_to_lds_b128) && \
    __has_builtin(__builtin_amdgcn_s_wait_asynccnt)
#define USE_ASYNC_LDS 1
typedef __attribute__((address_space(1))) i32x4 g_i32x4;
typedef __attribute__((address_space(3))) i32x4 l_i32x4;
#define ASYNC_CP_B128(gp, lp) \
  __builtin_amdgcn_global_load_async_to_lds_b128((g_i32x4*)(gp), (lp), 0, 0)
#else
#define USE_ASYNC_LDS 0
#endif

#if __has_builtin(__builtin_amdgcn_tanhf)
#define FAST_TANH(x) __builtin_amdgcn_tanhf(x)
#else
#define FAST_TANH(x) tanhf(x)
#endif

union Frag {
  v16bf v;
  u32x4 q[2];
};

// A-matrix 16x32 bf16 fragment (contiguous k order, same as the B loaders):
//   lanes 0-15 : M = lane,    v0..7 = K[kb..kb+15]
//   lanes16-31 : M = lane-16, v0..7 = K[kb+16..kb+31]
__device__ __forceinline__ v16bf load_a_frag(const bf16* base, int rowStride,
                                             int row0, int kb, int lane) {
  const int lo = lane & 15, hi = lane >> 4;
  const bf16* p = base + (size_t)(row0 + lo) * rowStride + kb + hi * 16;
  Frag f;
  f.q[0] = *(const u32x4*)(p);
  f.q[1] = *(const u32x4*)(p + 8);
  return f.v;
}

// B-matrix 32x16 bf16 fragment from LDS (row-major [n][k], stride LROW):
//   lanes 0-15 : N = lane,    K = kb..kb+15 ; lanes16-31 : N = lane-16, K = kb+16..kb+31
__device__ __forceinline__ v16bf load_b_frag_lds(const bf16* lds, int nLocal,
                                                 int kb, int lane) {
  const int lo = lane & 15, hi = lane >> 4;
  const bf16* p = lds + (nLocal + lo) * LROW + kb + hi * 16;
  Frag f;
  f.q[0] = *(const u32x4*)(p);
  f.q[1] = *(const u32x4*)(p + 8);
  return f.v;
}

// Same B fragment layout, direct from global row-major [n][k].
__device__ __forceinline__ v16bf load_b_frag_g(const bf16* base, int rowStride,
                                               int nrow0, int kb, int lane) {
  const int lo = lane & 15, hi = lane >> 4;
  const bf16* p = base + (size_t)(nrow0 + lo) * rowStride + kb + hi * 16;
  Frag f;
  f.q[0] = *(const u32x4*)(p);
  f.q[1] = *(const u32x4*)(p + 8);
  return f.v;
}

// --------------------------- prep kernels ----------------------------------

__global__ void k_prep_wcat(const float* __restrict__ wih, const unsigned char* __restrict__ imask,
                            const float* __restrict__ whh, const unsigned char* __restrict__ hmask,
                            bf16* __restrict__ Wcat) {
  const int total = HTOT * KCAT;
  for (int idx = blockIdx.x * blockDim.x + threadIdx.x; idx < total;
       idx += gridDim.x * blockDim.x) {
    const int n = idx / KCAT, k = idx - n * KCAT;
    float v;
    if (k < INTOT) {
      v = wih[(size_t)n * INTOT + k] * (imask[(size_t)n * INTOT + k] ? 1.0f : 0.0f);
    } else {
      const int k2 = k - INTOT;
      v = whh[(size_t)n * HTOT + k2] * (hmask[(size_t)n * HTOT + k2] ? 1.0f : 0.0f);
    }
    Wcat[idx] = (bf16)v;
  }
}

__global__ void k_prep_wout(const float* __restrict__ wout, const unsigned char* __restrict__ omask,
                            bf16* __restrict__ Woutm) {
  const int total = OUTTOT * HTOT;
  for (int idx = blockIdx.x * blockDim.x + threadIdx.x; idx < total;
       idx += gridDim.x * blockDim.x) {
    Woutm[idx] = (bf16)(wout[idx] * (omask[idx] ? 1.0f : 0.0f));
  }
}

__global__ void k_prep_bias(const float* __restrict__ bih, const float* __restrict__ bhh,
                            float* __restrict__ bias) {
  const int idx = blockIdx.x * blockDim.x + threadIdx.x;
  if (idx < HTOT) bias[idx] = bih[idx] + bhh[idx];
}

// fill x-columns of HX[t] for all t (rows = t*B+b flattened, col 0..511)
__global__ void k_prep_hx(const float* __restrict__ x, bf16* __restrict__ HX) {
  const int total = TSTEPS * BATCH * INTOT;
  for (int idx = blockIdx.x * blockDim.x + threadIdx.x; idx < total;
       idx += gridDim.x * blockDim.x) {
    const int row = idx >> 9;          // /512
    const int k   = idx & 511;
    HX[(size_t)row * KCAT + k] = (bf16)x[idx];
  }
}

// zero h-columns of HX[0] (h_{-1} = 0)
__global__ void k_zero_h0(bf16* __restrict__ HX) {
  const int total = BATCH * HTOT / 2;  // dword count
  unsigned int* p = (unsigned int*)HX;
  for (int idx = blockIdx.x * blockDim.x + threadIdx.x; idx < total;
       idx += gridDim.x * blockDim.x) {
    const int b  = idx >> 10;          // /1024 dwords per row
    const int c2 = idx & 1023;
    p[(size_t)b * (KCAT / 2) + (INTOT / 2) + c2] = 0u;
  }
}

// --------------------------- recurrence step -------------------------------
// grid = 32 blocks (N blocks of 64), 256 threads (8 waves).
// Wave w: M tile w (16 batch rows) x 4 N tiles. K loop: 40 uniform chunks of 64
// over HX[t] rows (stride KCAT); double-buffered LDS, 1 barrier/chunk.

__global__ __launch_bounds__(256) void k_step(bf16* __restrict__ HX,
                                              const bf16* __restrict__ Wcat,
                                              const float* __restrict__ bias,
                                              int t) {
  __shared__ __attribute__((aligned(16))) bf16 ldsB[2][NBLK * LROW];

  const int tid  = threadIdx.x;
  const int lane = tid & 31;
  const int wave = tid >> 5;
  const int lo   = lane & 15;
  const int hi   = lane >> 4;
  const int n0   = blockIdx.x * NBLK;  // N block base (hidden units)
  const int m0   = wave * 16;          // M tile base (batch rows)

  const bf16* At = HX + (size_t)t * BATCH * KCAT;                  // [x_t | h_{t-1}]
  bf16*       Hn = HX + (size_t)(t + 1) * BATCH * KCAT + INTOT;    // h_t columns

  // staging geometry: chunk = Wcat[n0..n0+63][c..c+63] = 512 x 16B segments.
  // thread stages rows (srow, srow+32), segment sseg.
  const int srow = tid >> 3;           // 0..31
  const int sseg = tid & 7;            // 0..7
  const bf16* g0 = Wcat + (size_t)(n0 + srow) * KCAT + sseg * 8;
  const bf16* g1 = Wcat + (size_t)(n0 + srow + 32) * KCAT + sseg * 8;

  const v8f vzero = {0.f, 0.f, 0.f, 0.f, 0.f, 0.f, 0.f, 0.f};
  v8f acc[NTILES];
#pragma unroll
  for (int j = 0; j < NTILES; ++j) acc[j] = vzero;

#if USE_ASYNC_LDS
  // pre-cast LDS destinations to AS(3) once (hoists addrspacecast null-guards)
  l_i32x4* l0p[2] = {(l_i32x4*)&ldsB[0][srow * LROW + sseg * 8],
                     (l_i32x4*)&ldsB[1][srow * LROW + sseg * 8]};
  l_i32x4* l1p[2] = {(l_i32x4*)&ldsB[0][(srow + 32) * LROW + sseg * 8],
                     (l_i32x4*)&ldsB[1][(srow + 32) * LROW + sseg * 8]};

  // prologue: chunk 0 -> buf 0 (async copy engine, no VGPR round-trip)
  ASYNC_CP_B128(g0, l0p[0]);
  ASYNC_CP_B128(g1, l1p[0]);
  for (int kc = 0; kc < NCHUNK; ++kc) {
    __builtin_amdgcn_s_wait_asynccnt(0);  // our chunk-kc copies landed
    __syncthreads();                      // everyone's landed; buf[nxt] is free
    if (kc + 1 < NCHUNK) {
      const int nxt = (kc + 1) & 1;
      ASYNC_CP_B128(g0 + (size_t)(kc + 1) * KCH, l0p[nxt]);
      ASYNC_CP_B128(g1 + (size_t)(kc + 1) * KCH, l1p[nxt]);
    }
    const bf16* buf = ldsB[kc & 1];
#pragma unroll
    for (int sub = 0; sub < 2; ++sub) {
      const v16bf a = load_a_frag(At, KCAT, m0, kc * KCH + sub * 32, lane);
#pragma unroll
      for (int j = 0; j < NTILES; ++j) {
        const v16bf b = load_b_frag_lds(buf, j * 16, sub * 32, lane);
        acc[j] = __builtin_amdgcn_wmma_f32_16x16x32_bf16(
            false, a, false, b, (short)0, acc[j], false, false);
      }
    }
  }
#else
  // fallback: global->VGPR->LDS double-buffered, 1 barrier/chunk
  bf16* l0[2] = {&ldsB[0][srow * LROW + sseg * 8], &ldsB[1][srow * LROW + sseg * 8]};
  bf16* l1[2] = {&ldsB[0][(srow + 32) * LROW + sseg * 8], &ldsB[1][(srow + 32) * LROW + sseg * 8]};
  u32x4 r0 = *(const u32x4*)(g0);
  u32x4 r1 = *(const u32x4*)(g1);
  *(u32x4*)l0[0] = r0;
  *(u32x4*)l1[0] = r1;
  r0 = *(const u32x4*)(g0 + KCH);
  r1 = *(const u32x4*)(g1 + KCH);
  __syncthreads();
  for (int kc = 0; kc < NCHUNK; ++kc) {
    if (kc + 1 < NCHUNK) {
      const int nxt = (kc + 1) & 1;
      *(u32x4*)l0[nxt] = r0;
      *(u32x4*)l1[nxt] = r1;
    }
    if (kc + 2 < NCHUNK) {
      r0 = *(const u32x4*)(g0 + (size_t)(kc + 2) * KCH);
      r1 = *(const u32x4*)(g1 + (size_t)(kc + 2) * KCH);
    }
    const bf16* buf = ldsB[kc & 1];
#pragma unroll
    for (int sub = 0; sub < 2; ++sub) {
      const v16bf a = load_a_frag(At, KCAT, m0, kc * KCH + sub * 32, lane);
#pragma unroll
      for (int j = 0; j < NTILES; ++j) {
        const v16bf b = load_b_frag_lds(buf, j * 16, sub * 32, lane);
        acc[j] = __builtin_amdgcn_wmma_f32_16x16x32_bf16(
            false, a, false, b, (short)0, acc[j], false, false);
      }
    }
    __syncthreads();
  }
#endif

  // epilogue: h = tanh(acc + bias), store bf16 into HX[t+1] h-columns
#pragma unroll
  for (int j = 0; j < NTILES; ++j) {
    const int ng = n0 + j * 16 + lo;   // hidden index (C/D: N = lane%16)
    const float bv = bias[ng];
#pragma unroll
    for (int r = 0; r < 8; ++r) {
      const int mg = m0 + r + hi * 8;  // batch row (C/D: M = vgpr + 8*(lane>=16))
      const float v = FAST_TANH(acc[j][r] + bv);
      Hn[(size_t)mg * KCAT + ng] = (bf16)v;
    }
  }
}

// --------------------------- readout ---------------------------------------
// out[t,m,b,o] = h_t[b,:] @ Wout_m^T + b_out ; block-diagonal (K = 512 per module).
// grid = 100 blocks x 256 threads; wave: one 16-row tile of (t*B+b) x 4 modules.

__global__ __launch_bounds__(256) void k_readout(const bf16* __restrict__ HX,
                                                 const bf16* __restrict__ Woutm,
                                                 const float* __restrict__ bout,
                                                 float* __restrict__ out) {
  const int tid  = threadIdx.x;
  const int lane = tid & 31;
  const int wave = tid >> 5;
  const int lo   = lane & 15;
  const int hi   = lane >> 4;

  const int mt = blockIdx.x * 8 + wave;  // 0..799 over flattened (t*B+b)/16
  const int g0 = mt * 16;
  const int t  = g0 >> 7;                // 16 | 128, tile never crosses t
  const int b0 = g0 & 127;
  // h_t lives in HX[t+1] columns INTOT..INTOT+HTOT
  const bf16* Abase = HX + (size_t)(t + 1) * BATCH * KCAT + INTOT;

  const v8f vzero = {0.f, 0.f, 0.f, 0.f, 0.f, 0.f, 0.f, 0.f};

  for (int m = 0; m < MMOD; ++m) {
    v8f acc = vzero;
    const int kbase = m * HID;
#pragma unroll 4
    for (int kc = 0; kc < (HID / 32); ++kc) {
      const v16bf a   = load_a_frag(Abase, KCAT, b0, kbase + kc * 32, lane);
      const v16bf bfr = load_b_frag_g(Woutm, HTOT, m * 16, kbase + kc * 32, lane);
      acc = __builtin_amdgcn_wmma_f32_16x16x32_bf16(
          false, a, false, bfr, (short)0, acc, false, false);
    }
    const int o = lo;
    const float bo = bout[m * 16 + o];
#pragma unroll
    for (int r = 0; r < 8; ++r) {
      const int brow = b0 + r + hi * 8;
      out[(((size_t)t * MMOD + m) * BATCH + brow) * 16 + o] = acc[r] + bo;
    }
  }
}

// --------------------------- host side -------------------------------------

extern "C" void kernel_launch(void* const* d_in, const int* in_sizes, int n_in,
                              void* d_out, int out_size, void* d_ws, size_t ws_size,
                              hipStream_t stream) {
  const float*         x     = (const float*)d_in[0];
  const float*         wih   = (const float*)d_in[1];
  const float*         bih   = (const float*)d_in[2];
  const float*         whh   = (const float*)d_in[3];
  const float*         bhh   = (const float*)d_in[4];
  const float*         wout  = (const float*)d_in[5];
  const float*         bout  = (const float*)d_in[6];
  const unsigned char* imask = (const unsigned char*)d_in[7];
  const unsigned char* hmask = (const unsigned char*)d_in[8];
  const unsigned char* omask = (const unsigned char*)d_in[9];

  char* ws = (char*)d_ws;
  size_t off = 0;
  bf16* Wcat  = (bf16*)(ws + off); off += (size_t)HTOT * KCAT * sizeof(bf16);
  bf16* Woutm = (bf16*)(ws + off); off += (size_t)OUTTOT * HTOT * sizeof(bf16);
  bf16* HX    = (bf16*)(ws + off); off += (size_t)(TSTEPS + 1) * BATCH * KCAT * sizeof(bf16);
  float* bias = (float*)(ws + off); off += (size_t)HTOT * sizeof(float);
  (void)ws_size; (void)in_sizes; (void)n_in; (void)out_size;

  k_prep_wcat<<<512, 256, 0, stream>>>(wih, imask, whh, hmask, Wcat);
  k_prep_wout<<<64, 256, 0, stream>>>(wout, omask, Woutm);
  k_prep_bias<<<8, 256, 0, stream>>>(bih, bhh, bias);
  k_prep_hx<<<1024, 256, 0, stream>>>(x, HX);
  k_zero_h0<<<128, 256, 0, stream>>>(HX);

  for (int t = 0; t < TSTEPS; ++t) {
    k_step<<<32, 256, 0, stream>>>(HX, Wcat, bias, t);
  }

  k_readout<<<100, 256, 0, stream>>>(HX, Woutm, bout, (float*)d_out);
}
